// HSR_2_25116968747549
// MI455X (gfx1250) — matrix operand order repaired and checked by reading, not compiled, not run on hardware
//
#include <hip/hip_runtime.h>
#include <hip/hip_bf16.h>

typedef __attribute__((ext_vector_type(16))) _Float16 v16h;
typedef __attribute__((ext_vector_type(8)))  _Float16 v8h;
typedef __attribute__((ext_vector_type(8)))  float    v8f;

#define NNODES 8192
#define DD     64
#define HH     4
#define HD     256
#define ENC_NEG_INF 0x007FFFFFu   // encode(-inf) in order-preserving uint space

// ---------- order-preserving float<->uint encoding for atomic max ----------
__device__ __forceinline__ unsigned int encF(float f) {
    unsigned int u = __float_as_uint(f);
    return (f >= 0.0f) ? (u | 0x80000000u) : ~u;
}
__device__ __forceinline__ float decF(unsigned int u) {
    return (u & 0x80000000u) ? __uint_as_float(u & 0x7FFFFFFFu) : __uint_as_float(~u);
}
__device__ __forceinline__ float lrelu(float v, float s) { return v > 0.0f ? v : v * s; }

// ---------------------- small utility kernels ----------------------
__global__ void fill_u32(unsigned int* __restrict__ p, unsigned int v, int n) {
    int i = blockIdx.x * blockDim.x + threadIdx.x;
    if (i < n) p[i] = v;
}

__global__ void cvt_f16(const float* __restrict__ in, _Float16* __restrict__ out, int n) {
    int i = blockIdx.x * blockDim.x + threadIdx.x;
    if (i < n) out[i] = (_Float16)in[i];
}

// out[c*rows + r] = in[r*cols + c]  (produce transposed f16 weights: Ncols x K)
__global__ void cvtT_f16(const float* __restrict__ in, _Float16* __restrict__ out, int rows, int cols) {
    int i = blockIdx.x * blockDim.x + threadIdx.x;
    if (i >= rows * cols) return;
    int r = i / cols, c = i - r * cols;
    out[(size_t)c * rows + r] = (_Float16)in[i];
}

// cbw[j] = sum_k cb[k] * linw[k*DD + j]   (fold bias through output projection)
__global__ void cbw_kernel(const float* __restrict__ cb, const float* __restrict__ linw,
                           float* __restrict__ outb) {
    int j = threadIdx.x;
    float s = 0.0f;
    for (int k = 0; k < HD; ++k) s += cb[k] * linw[k * DD + j];
    outb[j] = s;
}

// ---------------------- WMMA GEMM, 1x4 tile register blocking ----------------------
// C[M x Ncols] = A[M x K](f16,row-major) * Bt[Ncols x K](f16,row-major)^T + bias, act
// one wave32 per 16x64 output strip (4 WMMA n-tiles sharing one A fragment).
// K multiple of 32; M multiple of 16; Ncols multiple of 64.
__global__ __launch_bounds__(256) void wmma_gemm(
    const _Float16* __restrict__ A, const _Float16* __restrict__ Bt,
    const float* __restrict__ bias, float* __restrict__ C,
    int M, int Ncols, int K, int act, int numTiles)
{
    const int wave = (blockIdx.x * blockDim.x + threadIdx.x) >> 5;
    if (wave >= numTiles) return;               // wave-uniform exit (EXEC all-1s for WMMA)
    const int lane = threadIdx.x & 31;
    const int tilesN = Ncols >> 6;              // 64-wide strips
    const int tm = wave / tilesN;
    const int tn = wave - tm * tilesN;
    const int hi  = lane >> 4;                  // half-wave selector
    const int l15 = lane & 15;

    // A fragment: lane holds row M=l15; K = i + 16*hi (matches B)
    const _Float16* arow = A + (size_t)(tm * 16 + l15) * K + 16 * hi;
    // B fragments: lane holds column N=l15 of each 16-wide subtile; K = i + 16*hi
    const _Float16* brow0 = Bt + (size_t)(tn * 64 + 0  + l15) * K + 16 * hi;
    const _Float16* brow1 = Bt + (size_t)(tn * 64 + 16 + l15) * K + 16 * hi;
    const _Float16* brow2 = Bt + (size_t)(tn * 64 + 32 + l15) * K + 16 * hi;
    const _Float16* brow3 = Bt + (size_t)(tn * 64 + 48 + l15) * K + 16 * hi;

    v8f c0 = {}, c1 = {}, c2 = {}, c3 = {};
    for (int k0 = 0; k0 < K; k0 += 32) {
        __builtin_prefetch(arow + k0 + 64, 0, 1);   // -> global_prefetch_b8
        v8h alo = *(const v8h*)(arow + k0);         // K = k0 + 16*hi + [0..7]
        v8h ahi = *(const v8h*)(arow + k0 + 8);     // K = k0 + 16*hi + 8 + [0..7]
        v16h a = __builtin_shufflevector(alo, ahi, 0,1,2,3,4,5,6,7,8,9,10,11,12,13,14,15);

        v8h b0l = *(const v8h*)(brow0 + k0), b0h = *(const v8h*)(brow0 + k0 + 8);
        v8h b1l = *(const v8h*)(brow1 + k0), b1h = *(const v8h*)(brow1 + k0 + 8);
        v8h b2l = *(const v8h*)(brow2 + k0), b2h = *(const v8h*)(brow2 + k0 + 8);
        v8h b3l = *(const v8h*)(brow3 + k0), b3h = *(const v8h*)(brow3 + k0 + 8);
        v16h b0 = __builtin_shufflevector(b0l, b0h, 0,1,2,3,4,5,6,7,8,9,10,11,12,13,14,15);
        v16h b1 = __builtin_shufflevector(b1l, b1h, 0,1,2,3,4,5,6,7,8,9,10,11,12,13,14,15);
        v16h b2 = __builtin_shufflevector(b2l, b2h, 0,1,2,3,4,5,6,7,8,9,10,11,12,13,14,15);
        v16h b3 = __builtin_shufflevector(b3l, b3h, 0,1,2,3,4,5,6,7,8,9,10,11,12,13,14,15);

        c0 = __builtin_amdgcn_wmma_f32_16x16x32_f16(false, a, false, b0, (short)0, c0, false, false);
        c1 = __builtin_amdgcn_wmma_f32_16x16x32_f16(false, a, false, b1, (short)0, c1, false, false);
        c2 = __builtin_amdgcn_wmma_f32_16x16x32_f16(false, a, false, b2, (short)0, c2, false, false);
        c3 = __builtin_amdgcn_wmma_f32_16x16x32_f16(false, a, false, b3, (short)0, c3, false, false);
    }

    // epilogue: C/D layout M = j + 8*hi, N = lane&15 within each subtile
    v8f acc[4] = {c0, c1, c2, c3};
    #pragma unroll
    for (int s = 0; s < 4; ++s) {
        const int col = tn * 64 + s * 16 + l15;
        const float bv = bias ? bias[col] : 0.0f;
        #pragma unroll
        for (int j = 0; j < 8; ++j) {
            int row = tm * 16 + j + 8 * hi;
            float v = acc[s][j] + bv;
            if (act == 1) v = lrelu(v, 0.01f);
            C[(size_t)row * Ncols + col] = v;
        }
    }
}

// ---------------------- GATv2 edge kernels ----------------------
// logits[e,h] = sum_k att[h,k] * lrelu(xl[s,h,k] + xr[d,h,k], 0.2); atomic segment-max into mEnc
__global__ void edge_logits_kernel(const int* __restrict__ src, const int* __restrict__ dst,
                                   const float* __restrict__ xl, const float* __restrict__ xr,
                                   const float* __restrict__ att, float* __restrict__ logits,
                                   unsigned int* __restrict__ mEnc, int E)
{
    int e = blockIdx.x * blockDim.x + threadIdx.x;
    if (e >= E) return;
    int s = src[e], d = dst[e];
    const float4* pl = (const float4*)(xl + (size_t)s * HD);
    const float4* pr = (const float4*)(xr + (size_t)d * HD);
    const float4* pa = (const float4*)att;
    #pragma unroll
    for (int h = 0; h < HH; ++h) {
        float acc = 0.0f;
        #pragma unroll
        for (int q = 0; q < DD / 4; ++q) {
            float4 a = pl[h * 16 + q], b = pr[h * 16 + q], w = pa[h * 16 + q];
            acc += w.x * lrelu(a.x + b.x, 0.2f);
            acc += w.y * lrelu(a.y + b.y, 0.2f);
            acc += w.z * lrelu(a.z + b.z, 0.2f);
            acc += w.w * lrelu(a.w + b.w, 0.2f);
        }
        logits[(size_t)e * HH + h] = acc;
        atomicMax(&mEnc[(size_t)d * HH + h], encF(acc));
    }
}

// ex = exp(logit - m[dst]); denom[dst] += ex  (ex overwrites logits buffer)
__global__ void exp_denom_kernel(const int* __restrict__ dst, const unsigned int* __restrict__ mEnc,
                                 float* __restrict__ logits, float* __restrict__ denom, int EH)
{
    int t = blockIdx.x * blockDim.x + threadIdx.x;
    if (t >= EH) return;
    int e = t >> 2, h = t & 3;
    int d = dst[e];
    float ex = expf(logits[t] - decF(mEnc[d * HH + h]));
    logits[t] = ex;
    atomicAdd(&denom[d * HH + h], ex);
}

// out[dst,h,:] += (ex/denom[dst,h]) * xl[src,h,:]
__global__ void scatter_kernel(const int* __restrict__ src, const int* __restrict__ dst,
                               const float* __restrict__ xl, const float* __restrict__ ex,
                               const float* __restrict__ denom, float* __restrict__ out, int EH)
{
    int t = blockIdx.x * blockDim.x + threadIdx.x;
    if (t >= EH) return;
    int e = t >> 2, h = t & 3;
    int s = src[e], d = dst[e];
    float alpha = ex[t] / denom[d * HH + h];
    const float* p = xl + (size_t)s * HD + h * DD;
    float* o = out + (size_t)d * HD + h * DD;
    #pragma unroll 8
    for (int k = 0; k < DD; ++k) atomicAdd(o + k, alpha * p[k]);
}

// ---------------------- LayerNorm (wave32, 64-wide rows) ----------------------
__global__ __launch_bounds__(256) void layernorm_kernel(const float* __restrict__ in, float* __restrict__ out,
                                                        const float* __restrict__ gamma,
                                                        const float* __restrict__ beta, int Nrows)
{
    int row  = (blockIdx.x * blockDim.x + threadIdx.x) >> 5;
    int lane = threadIdx.x & 31;
    if (row >= Nrows) return;
    const float* p = in + (size_t)row * DD;
    float a = p[lane], b = p[lane + 32];
    float s = a + b;
    for (int off = 16; off; off >>= 1) s += __shfl_xor(s, off, 32);
    float mu = s * (1.0f / DD);
    float da = a - mu, db = b - mu;
    float vs = da * da + db * db;
    for (int off = 16; off; off >>= 1) vs += __shfl_xor(vs, off, 32);
    float inv = rsqrtf(vs * (1.0f / DD) + 1e-5f);
    out[(size_t)row * DD + lane]      = da * inv * gamma[lane]      + beta[lane];
    out[(size_t)row * DD + lane + 32] = db * inv * gamma[lane + 32] + beta[lane + 32];
}

// ---------------------- host-side helpers ----------------------
static inline void launch_gemm(const _Float16* A, const _Float16* Bt, const float* bias, float* C,
                               int M, int Ncols, int K, int act, hipStream_t s)
{
    int tiles = (M / 16) * (Ncols / 64);   // one wave per 16x64 strip
    int blocks = (tiles + 7) / 8;          // 8 waves per 256-thread block
    wmma_gemm<<<blocks, 256, 0, s>>>(A, Bt, bias, C, M, Ncols, K, act, tiles);
}

extern "C" void kernel_launch(void* const* d_in, const int* in_sizes, int n_in,
                              void* d_out, int out_size, void* d_ws, size_t ws_size,
                              hipStream_t stream)
{
    // inputs (setup_inputs order)
    const float* x     = (const float*)d_in[0];
    const float* Wl1   = (const float*)d_in[1];
    const float* bl1   = (const float*)d_in[2];
    const float* Wr1   = (const float*)d_in[3];
    const float* br1   = (const float*)d_in[4];
    const float* att1  = (const float*)d_in[5];
    const float* cb1   = (const float*)d_in[6];
    const float* linw1 = (const float*)d_in[7];
    const float* Wl2   = (const float*)d_in[8];
    const float* bl2   = (const float*)d_in[9];
    const float* Wr2   = (const float*)d_in[10];
    const float* br2   = (const float*)d_in[11];
    const float* att2  = (const float*)d_in[12];
    const float* cb2   = (const float*)d_in[13];
    const float* linw2 = (const float*)d_in[14];
    const float* w1    = (const float*)d_in[15];
    const float* b1    = (const float*)d_in[16];
    const float* w2    = (const float*)d_in[17];
    const float* b2    = (const float*)d_in[18];
    const float* gamma = (const float*)d_in[19];
    const float* beta  = (const float*)d_in[20];
    const int*   esrc  = (const int*)d_in[21];
    const int*   edst  = (const int*)d_in[22];
    const int E  = in_sizes[21];
    const int EH = E * HH;
    float* yout = (float*)d_out;

    // bump allocator over workspace
    size_t off = 0;
    auto alloc = [&](size_t bytes) -> void* {
        void* p = (char*)d_ws + off;
        off += (bytes + 255) & ~(size_t)255;
        return p;
    };
    _Float16* h_in64  = (_Float16*)alloc((size_t)NNODES * DD * 2);   // f16 input for K=64 GEMMs
    _Float16* h_acc   = (_Float16*)alloc((size_t)NNODES * HD * 2);   // f16 input for K=256 GEMM
    _Float16* WlT1h   = (_Float16*)alloc(HD * DD * 2);
    _Float16* WrT1h   = (_Float16*)alloc(HD * DD * 2);
    _Float16* linT1h  = (_Float16*)alloc(DD * HD * 2);
    _Float16* w1Th    = (_Float16*)alloc(DD * DD * 2);
    _Float16* WlT2h   = (_Float16*)alloc(HD * DD * 2);
    _Float16* WrT2h   = (_Float16*)alloc(HD * DD * 2);
    _Float16* linT2h  = (_Float16*)alloc(DD * HD * 2);
    _Float16* w2Th    = (_Float16*)alloc(DD * DD * 2);
    float* cbw1       = (float*)alloc(DD * 4);
    float* cbw2       = (float*)alloc(DD * 4);
    float* XL         = (float*)alloc((size_t)NNODES * HD * 4);
    float* XR         = (float*)alloc((size_t)NNODES * HD * 4);
    float* out_acc    = (float*)alloc((size_t)NNODES * HD * 4);
    float* logits     = (float*)alloc((size_t)EH * 4);
    unsigned int* mEnc= (unsigned int*)alloc((size_t)NNODES * HH * 4);
    float* denom      = (float*)alloc((size_t)NNODES * HH * 4);
    float* g          = (float*)alloc((size_t)NNODES * DD * 4);
    float* t          = (float*)alloc((size_t)NNODES * DD * 4);
    float* ln         = (float*)alloc((size_t)NNODES * DD * 4);
    (void)ws_size; (void)n_in; (void)out_size;

    const int T = 256;
    auto nb = [](int n) { return (n + 255) / 256; };

    // ---- one-time (per call) weight prep: transpose + f16, fold cb through linw ----
    cvtT_f16<<<nb(DD * HD), T, 0, stream>>>(Wl1,   WlT1h,  DD, HD);  // (K=64 x 256) -> (256 x 64)
    cvtT_f16<<<nb(DD * HD), T, 0, stream>>>(Wr1,   WrT1h,  DD, HD);
    cvtT_f16<<<nb(HD * DD), T, 0, stream>>>(linw1, linT1h, HD, DD);  // (256 x 64) -> (64 x 256)
    cvtT_f16<<<nb(DD * DD), T, 0, stream>>>(w1,    w1Th,   DD, DD);
    cvtT_f16<<<nb(DD * HD), T, 0, stream>>>(Wl2,   WlT2h,  DD, HD);
    cvtT_f16<<<nb(DD * HD), T, 0, stream>>>(Wr2,   WrT2h,  DD, HD);
    cvtT_f16<<<nb(HD * DD), T, 0, stream>>>(linw2, linT2h, HD, DD);
    cvtT_f16<<<nb(DD * DD), T, 0, stream>>>(w2,    w2Th,   DD, DD);
    cbw_kernel<<<1, DD, 0, stream>>>(cb1, linw1, cbw1);
    cbw_kernel<<<1, DD, 0, stream>>>(cb2, linw2, cbw2);

    // ---- GAT layer runner ----
    auto gat_layer = [&](const float* xin, const _Float16* WlT, const float* bl,
                         const _Float16* WrT, const float* br, const float* att,
                         const _Float16* linT, const float* cbw, float* gout) {
        cvt_f16<<<nb(NNODES * DD), T, 0, stream>>>(xin, h_in64, NNODES * DD);
        launch_gemm(h_in64, WlT, bl, XL, NNODES, HD, DD, 0, stream);       // xl = x@Wl + bl
        launch_gemm(h_in64, WrT, br, XR, NNODES, HD, DD, 0, stream);       // xr = x@Wr + br
        fill_u32<<<nb(NNODES * HH), T, 0, stream>>>(mEnc, ENC_NEG_INF, NNODES * HH);
        fill_u32<<<nb(NNODES * HH), T, 0, stream>>>((unsigned int*)denom, 0u, NNODES * HH);
        fill_u32<<<nb(NNODES * HD), T, 0, stream>>>((unsigned int*)out_acc, 0u, NNODES * HD);
        edge_logits_kernel<<<nb(E), T, 0, stream>>>(esrc, edst, XL, XR, att, logits, mEnc, E);
        exp_denom_kernel<<<nb(EH), T, 0, stream>>>(edst, mEnc, logits, denom, EH);
        scatter_kernel<<<nb(EH), T, 0, stream>>>(esrc, edst, XL, logits, denom, out_acc, EH);
        cvt_f16<<<nb(NNODES * HD), T, 0, stream>>>(out_acc, h_acc, NNODES * HD);
        launch_gemm(h_acc, linT, cbw, gout, NNODES, DD, HD, 0, stream);    // (out+cb)@linw
    };

    // ---- layer 1 ----
    gat_layer(x, WlT1h, bl1, WrT1h, br1, att1, linT1h, cbw1, g);
    cvt_f16<<<nb(NNODES * DD), T, 0, stream>>>(g, h_in64, NNODES * DD);
    launch_gemm(h_in64, w1Th, b1, t, NNODES, DD, DD, 1, stream);           // lrelu(g@w1+b1, 0.01)
    layernorm_kernel<<<nb(NNODES * 32), T, 0, stream>>>(t, ln, gamma, beta, NNODES);

    // ---- layer 2 ----
    gat_layer(ln, WlT2h, bl2, WrT2h, br2, att2, linT2h, cbw2, g);
    cvt_f16<<<nb(NNODES * DD), T, 0, stream>>>(g, h_in64, NNODES * DD);
    launch_gemm(h_in64, w2Th, b2, yout, NNODES, DD, DD, 1, stream);        // lrelu(g@w2+b2, 0.01)
}
